// JointNetwork_48438641164781
// MI455X (gfx1250) — compile-verified, matrix operand order checked by reading, not run on hardware
//
#include <hip/hip_runtime.h>

// ---- CDNA5 vector types ----
typedef _Float16 half8  __attribute__((ext_vector_type(8)));
typedef _Float16 v16h   __attribute__((ext_vector_type(16)));
typedef float    v8f    __attribute__((ext_vector_type(8)));
typedef float    f32x4  __attribute__((ext_vector_type(4)));

// Problem constants (from reference)
constexpr int Bn = 8;      // batch
constexpr int T  = 256;
constexpr int U  = 64;
constexpr int H  = 640;    // hidden (K)
constexpr int V  = 1024;   // vocab (N)

constexpr int KC   = 32;   // K chunk per WMMA step (f16 16x16x32)
constexpr int TN   = 128;  // v tile per block
constexpr int TMT  = 8;    // t rows per block
constexpr int TMU  = 16;   // u rows per block  -> 128 M rows per block
constexpr int LDSS = 40;   // padded LDS row stride in halves (80B: conflict-free b128 frags)
constexpr int LBUF = 128 * LDSS;  // halves per buffer

__global__ __launch_bounds__(256)
void joint_wmma_f16(const float* __restrict__ f,
                    const float* __restrict__ p,
                    const float* __restrict__ W,
                    const float* __restrict__ bias,
                    float* __restrict__ out) {
  // double-buffered staging: 2 x (Act + W) = 40KB of the 320KB WGP LDS
  __shared__ __align__(16) _Float16 sAct[2][LBUF];
  __shared__ __align__(16) _Float16 sW  [2][LBUF];

  const int tid   = threadIdx.x;
  const int wv    = tid >> 5;        // wave 0..7  -> 16-row M strip
  const int lane  = tid & 31;
  const int nlane = lane & 15;       // N column within 16x16 tile
  const int hi16  = lane >> 4;       // lane-group selector (k sub-chunk select)

  const int b  = blockIdx.z;
  const int v0 = blockIdx.x * TN;
  const int t0 = (blockIdx.y >> 2) * TMT;  // 32 t-blocks
  const int u0 = (blockIdx.y & 3) * TMU;   // 4 u-blocks

  // ---- accumulators (8 n-frags of 16x16 f32), initialized with bias ----
  // C/D layout: a lane's 8 elems all share N = lane&15 (M = g + 8*hi16)
  v8f acc[8];
#pragma unroll
  for (int j = 0; j < 8; ++j) {
    const float bv = bias[v0 + 16 * j + nlane];
#pragma unroll
    for (int g = 0; g < 8; ++g) acc[j][g] = bv;
  }

  // ---- staging thread mapping: 2 threads per row, 16 k each ----
  const int srow  = tid >> 1;          // 0..127
  const int skoff = (tid & 1) * 16;    // 0 or 16
  const float* fRow = f + (size_t)(b * T + (t0 + (srow >> 4))) * H + skoff;
  const float* pRow = p + (size_t)(b * U + (u0 + (srow & 15))) * H + skoff;
  const float* wRow = W + (size_t)(v0 + srow) * H + skoff;

  // ---- per-lane fragment addresses in LDS (halves) ----
  // A/B 16-bit frag: lanes 0-15 take k {0..7,16..23}; lanes 16-31 take {8..15,24..31}
  const int arow  = 16 * wv + nlane;
  const int aoff0 = arow * LDSS + hi16 * 8;
  const int aoff1 = arow * LDSS + 16 + hi16 * 8;

  // staged global registers for one K chunk (f, p, W: 16 floats each)
  f32x4 rf[4], rp[4], rw[4];

  auto loadChunk = [&](int kc) {
    const f32x4* f4 = (const f32x4*)(fRow + kc);
    const f32x4* p4 = (const f32x4*)(pRow + kc);
    const f32x4* w4 = (const f32x4*)(wRow + kc);
#pragma unroll
    for (int i = 0; i < 4; ++i) { rf[i] = f4[i]; rp[i] = p4[i]; rw[i] = w4[i]; }
  };

  loadChunk(0);
  int parity = 0;

  for (int kc = 0; kc < H; kc += KC) {
    // ---- convert staged registers -> f16 LDS tiles (buffer `parity`) ----
    {
      half8 h0, h1, g0, g1;
#pragma unroll
      for (int e = 0; e < 4; ++e) {
        float x0 = rf[0][e] + rp[0][e];
        float x1 = rf[1][e] + rp[1][e];
        float x2 = rf[2][e] + rp[2][e];
        float x3 = rf[3][e] + rp[3][e];
        h0[e]     = (_Float16)fmaxf(x0, 0.0f);
        h0[4 + e] = (_Float16)fmaxf(x1, 0.0f);
        h1[e]     = (_Float16)fmaxf(x2, 0.0f);
        h1[4 + e] = (_Float16)fmaxf(x3, 0.0f);
        g0[e]     = (_Float16)rw[0][e];
        g0[4 + e] = (_Float16)rw[1][e];
        g1[e]     = (_Float16)rw[2][e];
        g1[4 + e] = (_Float16)rw[3][e];
      }
      _Float16* adst = &sAct[parity][srow * LDSS + skoff];
      *(half8*)(adst)     = h0;
      *(half8*)(adst + 8) = h1;
      _Float16* wdst = &sW[parity][srow * LDSS + skoff];
      *(half8*)(wdst)     = g0;
      *(half8*)(wdst + 8) = g1;
    }

    __syncthreads();  // one barrier per iteration (double-buffered LDS)

    // ---- issue next chunk's global loads AFTER the barrier: the barrier
    // then only needs a dscnt wait, and these loads (L2-resident inputs)
    // stay in flight across the whole WMMA chain below, absorbed by the
    // s_wait_loadcnt at the NEXT iteration's convert phase. ----
    if (kc + KC < H) loadChunk(kc + KC);

    // ---- A fragment for this wave's 16-row strip ----
    half8 alo = *(const half8*)&sAct[parity][aoff0];
    half8 ahi = *(const half8*)&sAct[parity][aoff1];
    v16h afrag = __builtin_shufflevector(alo, ahi,
        0, 1, 2, 3, 4, 5, 6, 7, 8, 9, 10, 11, 12, 13, 14, 15);

    // ---- preload all 8 B fragments (lets scheduler overlap LDS & WMMA) ----
    v16h bfr[8];
#pragma unroll
    for (int j = 0; j < 8; ++j) {
      const int brow = 16 * j + nlane;
      half8 blo = *(const half8*)&sW[parity][brow * LDSS + hi16 * 8];
      half8 bhi = *(const half8*)&sW[parity][brow * LDSS + 16 + hi16 * 8];
      bfr[j] = __builtin_shufflevector(blo, bhi,
          0, 1, 2, 3, 4, 5, 6, 7, 8, 9, 10, 11, 12, 13, 14, 15);
    }

    // ---- 8 WMMAs across the 128-wide v tile ----
#pragma unroll
    for (int j = 0; j < 8; ++j) {
      acc[j] = __builtin_amdgcn_wmma_f32_16x16x32_f16(
          /*neg_a=*/false, afrag, /*neg_b=*/false, bfr[j],
          /*c_mod=*/(short)0, acc[j], /*reuse_a=*/false, /*reuse_b=*/false);
    }

    parity ^= 1;
  }

  // ---- epilogue: out[b,t,u,v]; per-lane rows M = g + 8*hi16 -> u, N -> v ----
  const int tt = t0 + wv;
#pragma unroll
  for (int j = 0; j < 8; ++j) {
    const int vv = v0 + 16 * j + nlane;
#pragma unroll
    for (int g = 0; g < 8; ++g) {
      const int uu = u0 + g + 8 * hi16;
      out[((size_t)(b * T + tt) * U + uu) * V + vv] = acc[j][g];
    }
  }
}

extern "C" void kernel_launch(void* const* d_in, const int* in_sizes, int n_in,
                              void* d_out, int out_size, void* d_ws, size_t ws_size,
                              hipStream_t stream) {
  (void)in_sizes; (void)n_in; (void)d_ws; (void)ws_size; (void)out_size;
  const float* f    = (const float*)d_in[0];
  const float* p    = (const float*)d_in[1];
  const float* W    = (const float*)d_in[2];
  const float* bias = (const float*)d_in[3];
  float* out = (float*)d_out;

  dim3 grid(V / TN, (T / TMT) * (U / TMU), Bn);  // (8, 128, 8)
  dim3 block(256);
  joint_wmma_f16<<<grid, block, 0, stream>>>(f, p, W, bias, out);
}
